// LSTMReactor_51651276702086
// MI455X (gfx1250) — compile-verified, matrix operand order checked by reading, not run on hardware
//
#include <hip/hip_runtime.h>
#include <hip/hip_bf16.h>

// ---------------------------------------------------------------------------
// CDNA5 / gfx1250: WMMA f16 GEMMs with fragment-packed weights (all B-side
// loads are b128), TDM (tensor_load_to_lds) staging of A tiles, persistent
// single-WGP LSTM recurrence (register-blocked 4 N-tiles/wave, h/c in LDS).
// ---------------------------------------------------------------------------

typedef __attribute__((ext_vector_type(16))) _Float16 v16h;
typedef __attribute__((ext_vector_type(8)))  float    v8f;
typedef __attribute__((ext_vector_type(4)))  unsigned int u32x4;
typedef __attribute__((ext_vector_type(8)))  int      i32x8;
typedef __attribute__((ext_vector_type(4)))  int      i32x4;

// Load 16 contiguous halves (32B) as one WMMA operand: two b128 loads.
__device__ __forceinline__ v16h load_frag32B(const _Float16* p) {
    union { uint4 q[2]; v16h h; } u;
    u.q[0] = *(const uint4*)p;
    u.q[1] = *(const uint4*)(p + 8);
    return u.h;
}

// ---- TDM: DMA a 2-D f16 tile (tile_h x tile_w elems, row stride in elems)
// from global memory into LDS at byte offset lds_off. D# layout per ISA §8.
__device__ __forceinline__ void tdm_load_2d_f16(unsigned lds_off, const void* gaddr,
                                                unsigned tile_w, unsigned tile_h,
                                                unsigned row_stride,
                                                unsigned tensor_w, unsigned tensor_h) {
    unsigned long long ga = (unsigned long long)gaddr;
    u32x4 g0;
    g0[0] = 1u;                                            // count=1 (valid user D#)
    g0[1] = lds_off;                                       // lds_addr (bytes)
    g0[2] = (unsigned)(ga & 0xffffffffu);                  // global_addr[31:0]
    g0[3] = (unsigned)((ga >> 32) & 0x01ffffffu) | (2u << 30); // addr[56:32] | type=2
    i32x8 g1;
    g1[0] = (int)(1u << 16);                               // data_size=1 (2 bytes)
    g1[1] = (int)((tensor_w & 0xffffu) << 16);             // tensor_dim0[15:0]
    g1[2] = (int)(((tensor_w >> 16) & 0xffffu) |           // tensor_dim0[31:16]
                  ((tensor_h & 0xffffu) << 16));           // tensor_dim1[15:0]
    g1[3] = (int)(((tensor_h >> 16) & 0xffffu) |           // tensor_dim1[31:16]
                  ((tile_w & 0xffffu) << 16));             // tile_dim0
    g1[4] = (int)(tile_h & 0xffffu);                       // tile_dim1 ; tile_dim2=0
    g1[5] = (int)row_stride;                               // tensor_dim0_stride[31:0]
    g1[6] = 0;                                             // stride[47:32], dim1_stride lo
    g1[7] = 0;
    i32x4 g2 = {0, 0, 0, 0};
    i32x4 g3 = {0, 0, 0, 0};
#if defined(__clang_major__) && (__clang_major__ >= 23)
    i32x8 g4 = {0, 0, 0, 0, 0, 0, 0, 0};
    __builtin_amdgcn_tensor_load_to_lds(g0, g1, g2, g3, g4, 0);
#else
    __builtin_amdgcn_tensor_load_to_lds(g0, g1, g2, g3, 0);
#endif
}

// ---- fp32 -> fp16 cast (activations) ---------------------------------------
__global__ void cast_f32_f16_kernel(const float* __restrict__ src, _Float16* __restrict__ dst,
                                    size_t n) {
    size_t i = (size_t)blockIdx.x * blockDim.x + threadIdx.x;
    if (i < n) dst[i] = (_Float16)src[i];
}

// ---- pack a KxN f32 weight matrix into WMMA B-fragment order (f16) ---------
__global__ void pack_b_kernel(const float* __restrict__ B, _Float16* __restrict__ P,
                              int K, int N) {
    int ktiles = K >> 5;
    int total  = (N >> 4) * ktiles * 32;
    int idx = blockIdx.x * blockDim.x + threadIdx.x;
    if (idx >= total) return;
    int lane = idx & 31;
    int kt   = (idx >> 5) % ktiles;
    int nt   = (idx >> 5) / ktiles;
    int col = lane & 15, khalf = lane >> 4;
    int k0 = kt * 32, n0 = nt * 16;
    _Float16* dst = P + (size_t)idx * 16;
#pragma unroll
    for (int i = 0; i < 16; ++i)
        dst[i] = (_Float16)B[(size_t)(k0 + khalf * 16 + i) * N + n0 + col];
}

// ---- Generic WMMA GEMM: C[MxN] = A[MxK](f16) * Bp(packed f16) --------------
// block = 256 threads = 8 waves; A tile staged into LDS by the TDM.
__global__ void wmma_gemm_kernel(const _Float16* __restrict__ A, const _Float16* __restrict__ Bp,
                                 const float* __restrict__ bias, float* __restrict__ C,
                                 int M, int N, int K, int fuse_lrelu) {
    __shared__ __align__(16) _Float16 sA[16 * 64];
    const int lane = threadIdx.x & 31;
    const int wave = threadIdx.x >> 5;          // 0..7
    const int m0   = blockIdx.y * 16;
    const int n0   = (blockIdx.x * 8 + wave) * 16;
    const bool active = (n0 < N);
    const int ktiles  = K >> 5;
    const _Float16* bp = Bp + ((size_t)(n0 >> 4) * ktiles * 32 + lane) * 16;
    const int arow = (lane & 15) * 64 + (lane >> 4) * 16;

    v8f acc = {};
    for (int k0 = 0; k0 < K; k0 += 64) {
        // Async DMA of the 16x64 A tile into LDS (wave 0 issues, all waves use).
        if (threadIdx.x < 32) {
            tdm_load_2d_f16(0u, &A[(size_t)m0 * K + k0],
                            64u, 16u, (unsigned)K, (unsigned)K, 16u);
            __builtin_amdgcn_s_wait_tensorcnt(0);
        }
        __syncthreads();
        if (active) {
            int kt = k0 >> 5;
            if (k0 + 64 < K)
                __builtin_prefetch(bp + (size_t)(kt + 2) * 512, 0, 0);
            v16h a0 = load_frag32B(&sA[arow]);
            v16h a1 = load_frag32B(&sA[arow + 32]);
            v16h b0 = load_frag32B(bp + (size_t)kt * 512);
            v16h b1 = load_frag32B(bp + (size_t)(kt + 1) * 512);
            acc = __builtin_amdgcn_wmma_f32_16x16x32_f16(false, a0, false, b0,
                                                         (short)0, acc, false, false);
            acc = __builtin_amdgcn_wmma_f32_16x16x32_f16(false, a1, false, b1,
                                                         (short)0, acc, false, false);
        }
        __syncthreads();
    }
    if (active) {
        const int col   = n0 + (lane & 15);
        const int rbase = m0 + ((lane >> 4) << 3);
        const float bv  = bias ? bias[col] : 0.0f;
#pragma unroll
        for (int v = 0; v < 8; ++v) {
            float val = acc[v] + bv;
            if (fuse_lrelu) val = (val > 0.0f) ? val : 0.2f * val;
            C[(size_t)(rbase + v) * N + col] = val;
        }
    }
}

// ---- in-place row LayerNorm with gain --------------------------------------
__global__ void ln_rows_kernel(float* __restrict__ X, const float* __restrict__ g, int D) {
    __shared__ float red[2];
    if (threadIdx.x == 0) { red[0] = 0.f; red[1] = 0.f; }
    __syncthreads();
    float* row = X + (size_t)blockIdx.x * D;
    float s = 0.f, s2 = 0.f;
    for (int i = threadIdx.x; i < D; i += blockDim.x) { float v = row[i]; s += v; s2 += v * v; }
    atomicAdd(&red[0], s);
    atomicAdd(&red[1], s2);
    __syncthreads();
    float mu  = red[0] / D;
    float var = red[1] / D - mu * mu;
    float rs  = rsqrtf(var + 1e-5f);
    for (int i = threadIdx.x; i < D; i += blockDim.x)
        row[i] = (row[i] - mu) * rs * g[i];
}

// ---- Persistent LSTM layer: one workgroup (32 waves), h/c resident in LDS --
// Rp packed fragments; each wave owns 4 N-tiles and shares the A fragment.
__global__ void lstm_layer_kernel(const float* __restrict__ wx,
                                  const _Float16* __restrict__ Rp,
                                  const float* __restrict__ bg,   // [2048]
                                  const float* __restrict__ g2,   // [2048]
                                  const float* __restrict__ g3,   // [512]
                                  const float* __restrict__ h0, int h0_stride,
                                  float* __restrict__ hs_out,     // (b*S+t)*512
                                  float* __restrict__ cs_out,
                                  int S) {
    extern __shared__ char smem_raw[];
    _Float16* h16 = (_Float16*)smem_raw;                           // 16*512 f16
    float*    c_s = (float*)(smem_raw + 16 * 512 * 2);             // 16*512 f32
    float*    v_s = c_s + 16 * 512;                                // 16*2048 f32
    float*    red = v_s + 16 * 2048;                               // 64 f32

    const int tid  = threadIdx.x;       // 0..1023
    const int lane = tid & 31;
    const int wave = tid >> 5;          // 0..31
    const int bIdx = tid >> 6;          // 0..15 : batch row (64 threads each)
    const int sub  = tid & 63;
    const int arow = (lane & 15) * 512 + (lane >> 4) * 16;

    const _Float16* bp0 = Rp + ((size_t)(wave +  0) * 16 * 32 + lane) * 16;
    const _Float16* bp1 = Rp + ((size_t)(wave + 32) * 16 * 32 + lane) * 16;
    const _Float16* bp2 = Rp + ((size_t)(wave + 64) * 16 * 32 + lane) * 16;
    const _Float16* bp3 = Rp + ((size_t)(wave + 96) * 16 * 32 + lane) * 16;

    for (int i = tid; i < 16 * 512; i += 1024) {
        int b_ = i >> 9, j = i & 511;
        float hv = h0[(size_t)b_ * h0_stride + j];
        h16[i] = (_Float16)hv;
        c_s[i] = hv;
    }
    __syncthreads();

    for (int t = 0; t < S; ++t) {
        if (tid < 32) red[tid] = 0.f;
        __syncthreads();

        // (1) v = h @ R : 4 register-blocked N-tiles per wave, shared A frag
        {
            v8f acc0 = {}, acc1 = {}, acc2 = {}, acc3 = {};
#pragma unroll 4
            for (int kt = 0; kt < 16; ++kt) {
                v16h a = load_frag32B(&h16[arow + kt * 32]);      // 2x ds_load_b128
                v16h b0 = load_frag32B(bp0 + (size_t)kt * 512);
                v16h b1 = load_frag32B(bp1 + (size_t)kt * 512);
                v16h b2 = load_frag32B(bp2 + (size_t)kt * 512);
                v16h b3 = load_frag32B(bp3 + (size_t)kt * 512);
                acc0 = __builtin_amdgcn_wmma_f32_16x16x32_f16(false, a, false, b0,
                                                              (short)0, acc0, false, false);
                acc1 = __builtin_amdgcn_wmma_f32_16x16x32_f16(false, a, false, b1,
                                                              (short)0, acc1, false, false);
                acc2 = __builtin_amdgcn_wmma_f32_16x16x32_f16(false, a, false, b2,
                                                              (short)0, acc2, false, false);
                acc3 = __builtin_amdgcn_wmma_f32_16x16x32_f16(false, a, false, b3,
                                                              (short)0, acc3, false, false);
            }
            int cl    = lane & 15;
            int rbase = (lane >> 4) << 3;
#pragma unroll
            for (int v = 0; v < 8; ++v) {
                int r = (rbase + v) * 2048;
                v_s[r + (wave +  0) * 16 + cl] = acc0[v];
                v_s[r + (wave + 32) * 16 + cl] = acc1[v];
                v_s[r + (wave + 64) * 16 + cl] = acc2[v];
                v_s[r + (wave + 96) * 16 + cl] = acc3[v];
            }
        }
        __syncthreads();

        // (2) LayerNorm over each row (2048) + wx_t + bias
        {
            float s = 0.f, s2 = 0.f;
            for (int i = sub; i < 2048; i += 64) {
                float v = v_s[bIdx * 2048 + i];
                s += v; s2 += v * v;
            }
            atomicAdd(&red[bIdx], s);
            atomicAdd(&red[16 + bIdx], s2);
        }
        __syncthreads();
        {
            float mu  = red[bIdx] * (1.f / 2048.f);
            float var = red[16 + bIdx] * (1.f / 2048.f) - mu * mu;
            float rs  = rsqrtf(var + 1e-5f);
            const float* wrow = wx + ((size_t)bIdx * S + t) * 2048;
            for (int i = sub; i < 2048; i += 64) {
                float v = v_s[bIdx * 2048 + i];
                v_s[bIdx * 2048 + i] = (v - mu) * rs * g2[i] + wrow[i] + bg[i];
            }
        }
        __syncthreads();
        if (tid < 32) red[tid] = 0.f;
        __syncthreads();

        // (3) gates + c update + c-LN partial sums
        float cn[8];
        {
            float s = 0.f, s2 = 0.f;
#pragma unroll
            for (int e = 0; e < 8; ++e) {
                int col  = sub + e * 64;
                float iv = v_s[bIdx * 2048 + col];
                float gv = v_s[bIdx * 2048 + 512 + col];
                float fv = v_s[bIdx * 2048 + 1024 + col];
                float co = c_s[bIdx * 512 + col];
                float si = 1.f / (1.f + __expf(-iv));
                float sf = 1.f / (1.f + __expf(-fv));
                float cnew = sf * co + si * tanhf(gv);
                cn[e] = cnew;
                c_s[bIdx * 512 + col] = cnew;
                cs_out[((size_t)bIdx * S + t) * 512 + col] = cnew;
                s += cnew; s2 += cnew * cnew;
            }
            atomicAdd(&red[bIdx], s);
            atomicAdd(&red[16 + bIdx], s2);
        }
        __syncthreads();

        // (4) h = sigmoid(o) * tanh(LN(c)*g3)
        {
            float mu  = red[bIdx] * (1.f / 512.f);
            float var = red[16 + bIdx] * (1.f / 512.f) - mu * mu;
            float rs  = rsqrtf(var + 1e-5f);
#pragma unroll
            for (int e = 0; e < 8; ++e) {
                int col  = sub + e * 64;
                float ov = v_s[bIdx * 2048 + 1536 + col];
                float so = 1.f / (1.f + __expf(-ov));
                float hv = so * tanhf((cn[e] - mu) * rs * g3[col]);
                h16[bIdx * 512 + col] = (_Float16)hv;
                hs_out[((size_t)bIdx * S + t) * 512 + col] = hv;
            }
        }
        __syncthreads();
    }
}

// ---- per-column mean/var over M rows (stride N) ----------------------------
__global__ void colstats_kernel(const float* __restrict__ X, float* __restrict__ mu,
                                float* __restrict__ var, int M, int N) {
    int col = blockIdx.x;
    __shared__ float red[2];
    if (threadIdx.x == 0) { red[0] = 0.f; red[1] = 0.f; }
    __syncthreads();
    float s = 0.f, s2 = 0.f;
    for (int r = threadIdx.x; r < M; r += blockDim.x) {
        float v = X[(size_t)r * N + col];
        s += v; s2 += v * v;
    }
    atomicAdd(&red[0], s);
    atomicAdd(&red[1], s2);
    __syncthreads();
    if (threadIdx.x == 0) {
        float m_ = red[0] / M;
        mu[col]  = m_;
        var[col] = red[1] / M - m_ * m_;
    }
}

// ---- BN (per-col stats) then per-row RMS scale, cast to f16 ----------------
__global__ void bn_rms_cast_kernel(const float* __restrict__ X, const float* __restrict__ mu,
                                   const float* __restrict__ var, const float* __restrict__ g,
                                   const float* __restrict__ b, _Float16* __restrict__ out,
                                   int D) {
    __shared__ float red;
    if (threadIdx.x == 0) red = 0.f;
    __syncthreads();
    const float* x = X + (size_t)blockIdx.x * D;
    float s2 = 0.f;
    for (int i = threadIdx.x; i < D; i += blockDim.x) {
        float v = (x[i] - mu[i]) * rsqrtf(var[i] + 1e-5f) * g[i] + b[i];
        s2 += v * v;
    }
    atomicAdd(&red, s2);
    __syncthreads();
    float scale = rsqrtf(red / D + 1e-8f);
    for (int i = threadIdx.x; i < D; i += blockDim.x) {
        float v = (x[i] - mu[i]) * rsqrtf(var[i] + 1e-5f) * g[i] + b[i];
        out[(size_t)blockIdx.x * D + i] = (_Float16)(v * scale);
    }
}

// ---- elementwise BN apply + cast to f16 ------------------------------------
__global__ void bn_cast_kernel(const float* __restrict__ X, const float* __restrict__ mu,
                               const float* __restrict__ var, const float* __restrict__ g,
                               const float* __restrict__ b, _Float16* __restrict__ out,
                               size_t total, int D) {
    size_t i = (size_t)blockIdx.x * blockDim.x + threadIdx.x;
    if (i >= total) return;
    int col = (int)(i % D);
    float v = (X[i] - mu[col]) * rsqrtf(var[col] + 1e-5f) * g[col] + b[col];
    out[i] = (_Float16)v;
}

// ---- per-k BN apply + transpose write into lw[(n*NS + k)*D + e] ------------
__global__ void a_bn_transpose_kernel(const float* __restrict__ T, const float* __restrict__ mu,
                                      const float* __restrict__ var, const float* __restrict__ g,
                                      const float* __restrict__ b, float* __restrict__ out,
                                      int k, int NS_, int D, size_t total) {
    size_t i = (size_t)blockIdx.x * blockDim.x + threadIdx.x;
    if (i >= total) return;
    size_t n = i / D;
    int e = (int)(i % D);
    float v = (T[i] - mu[e]) * rsqrtf(var[e] + 1e-5f) * g[e] + b[e];
    out[(n * NS_ + k) * D + e] = v;
}

// ---------------------------------------------------------------------------
extern "C" void kernel_launch(void* const* d_in, const int* in_sizes, int n_in,
                              void* d_out, int out_size, void* d_ws, size_t ws_size,
                              hipStream_t stream) {
    (void)in_sizes; (void)n_in; (void)out_size; (void)ws_size;

    const float* x     = (const float*)d_in[0];
    const float* m     = (const float*)d_in[1];
    const float* W0    = (const float*)d_in[2];
    const float* Wr    = (const float*)d_in[3];
    const float* R     = (const float*)d_in[4];
    const float* bb    = (const float*)d_in[5];
    const float* g1    = (const float*)d_in[6];
    const float* g2    = (const float*)d_in[7];
    const float* g3    = (const float*)d_in[8];
    const float* bn1_g = (const float*)d_in[9];
    const float* bn1_b = (const float*)d_in[10];
    const float* t_w1  = (const float*)d_in[11];
    const float* t_b1  = (const float*)d_in[12];
    const float* t_w2  = (const float*)d_in[13];
    const float* t_b2  = (const float*)d_in[14];
    const float* t_w3  = (const float*)d_in[15];
    const float* t_b3  = (const float*)d_in[16];
    const float* t_w4  = (const float*)d_in[17];
    const float* t_b4  = (const float*)d_in[18];
    const float* bn2_g = (const float*)d_in[19];
    const float* bn2_b = (const float*)d_in[20];
    const float* A_w   = (const float*)d_in[21];
    const float* A_b   = (const float*)d_in[22];
    const float* A_g   = (const float*)d_in[23];
    const float* A_be  = (const float*)d_in[24];

    constexpr int Bc = 16, Sc = 512, INc = 128, Hc = 512, Lc = 4, NSc = 18;
    constexpr int M  = Bc * Sc;       // 8192
    constexpr int G4 = 4 * Hc;        // 2048

    float* out     = (float*)d_out;
    float* lw      = out;                                        // [n][NS][512]
    float* inter_l = out + (size_t)M * NSc * Hc;                 // [L][b][t][512]
    float* inter_c = inter_l + (size_t)Lc * M * Hc;

    // ---- workspace carve ----
    char* wp = (char*)d_ws;
    auto carve = [&](size_t bytes) -> void* {
        void* p = wp;
        wp += (bytes + 255) & ~(size_t)255;
        return p;
    };
    _Float16* x16   = (_Float16*)carve((size_t)M * INc * 2);
    _Float16* W0p   = (_Float16*)carve((size_t)INc * G4 * 2);
    _Float16* Wrp   = (_Float16*)carve((size_t)(Lc - 1) * Hc * G4 * 2);
    _Float16* Rp    = (_Float16*)carve((size_t)Lc * Hc * G4 * 2);
    _Float16* w1p   = (_Float16*)carve((size_t)512 * 64 * 2);
    _Float16* w2p   = (_Float16*)carve((size_t)64 * 128 * 2);
    _Float16* w3p   = (_Float16*)carve((size_t)128 * 256 * 2);
    _Float16* w4p   = (_Float16*)carve((size_t)256 * 512 * 2);
    _Float16* Awp   = (_Float16*)carve((size_t)NSc * 512 * 512 * 2);
    _Float16* actA  = (_Float16*)carve((size_t)M * 512 * 2);
    _Float16* actB  = (_Float16*)carve((size_t)M * 512 * 2);
    float*    preact = (float*)carve((size_t)M * G4 * 4);
    float*    f32buf = (float*)carve((size_t)M * 512 * 4);
    float*    stats  = (float*)carve((size_t)4 * 512 * 4);

    auto cast = [&](const float* s, _Float16* d, size_t n) {
        cast_f32_f16_kernel<<<dim3((unsigned)((n + 255) / 256)), 256, 0, stream>>>(s, d, n);
    };
    auto pack = [&](const float* s, _Float16* d, int K, int N) {
        int total = (N >> 4) * (K >> 5) * 32;
        pack_b_kernel<<<(total + 255) / 256, 256, 0, stream>>>(s, d, K, N);
    };
    auto gemm = [&](const _Float16* A_, const _Float16* Bp_, const float* bias, float* C_,
                    int Mm, int Nn, int Kk, int lrelu) {
        dim3 grid((Nn + 127) / 128, Mm / 16);
        wmma_gemm_kernel<<<grid, 256, 0, stream>>>(A_, Bp_, bias, C_, Mm, Nn, Kk, lrelu);
    };

    // ---- cast activations, pack weights into fragment order ----
    cast(x, x16, (size_t)M * INc);
    pack(W0, W0p, INc, G4);
    for (int l = 0; l < Lc - 1; ++l)
        pack(Wr + (size_t)l * Hc * G4, Wrp + (size_t)l * Hc * G4, Hc, G4);
    for (int l = 0; l < Lc; ++l)
        pack(R + (size_t)l * Hc * G4, Rp + (size_t)l * Hc * G4, Hc, G4);
    pack(t_w1, w1p, 512, 64);
    pack(t_w2, w2p, 64, 128);
    pack(t_w3, w3p, 128, 256);
    pack(t_w4, w4p, 256, 512);
    for (int k = 0; k < NSc; ++k)
        pack(A_w + (size_t)k * 512 * 512, Awp + (size_t)k * 512 * 512, 512, 512);

    // ---- stacked LN-LSTM ----
    const _Float16* lin = x16;
    int Kin = INc;
    const float* h0 = m;
    int h0_stride = Hc;
    size_t lstm_smem = (size_t)16 * 512 * 2 + ((size_t)16 * 512 + 16 * 2048 + 64) * 4;

    for (int li = 0; li < Lc; ++li) {
        const _Float16* Wlp = (li == 0) ? W0p : (Wrp + (size_t)(li - 1) * Hc * G4);
        gemm(lin, Wlp, nullptr, preact, M, G4, Kin, 0);
        ln_rows_kernel<<<M, 256, 0, stream>>>(preact, g1 + (size_t)li * G4, G4);

        float* hs_l = inter_l + (size_t)li * M * Hc;
        float* cs_l = inter_c + (size_t)li * M * Hc;
        lstm_layer_kernel<<<1, 1024, (unsigned)lstm_smem, stream>>>(
            preact, Rp + (size_t)li * Hc * G4,
            bb + (size_t)li * G4, g2 + (size_t)li * G4, g3 + (size_t)li * Hc,
            h0, h0_stride, hs_l, cs_l, Sc);

        cast(hs_l, actA, (size_t)M * Hc);
        lin = actA;
        Kin = Hc;
        h0 = hs_l + (size_t)(Sc - 1) * Hc;
        h0_stride = Sc * Hc;
    }

    // ---- trunk: BN1 -> RMS -> 4x (GEMM + lrelu) -> BN2 ----
    float* hs_last = inter_l + (size_t)(Lc - 1) * M * Hc;
    float* mu1 = stats;        float* var1 = stats + 512;
    float* mu2 = stats + 1024; float* var2 = stats + 1536;

    colstats_kernel<<<512, 256, 0, stream>>>(hs_last, mu1, var1, M, Hc);
    bn_rms_cast_kernel<<<M, 256, 0, stream>>>(hs_last, mu1, var1, bn1_g, bn1_b, actA, Hc);

    gemm(actA, w1p, t_b1, f32buf, M, 64, 512, 1);
    cast(f32buf, actB, (size_t)M * 64);
    gemm(actB, w2p, t_b2, f32buf, M, 128, 64, 1);
    cast(f32buf, actA, (size_t)M * 128);
    gemm(actA, w3p, t_b3, f32buf, M, 256, 128, 1);
    cast(f32buf, actB, (size_t)M * 256);
    gemm(actB, w4p, t_b4, f32buf, M, 512, 256, 1);

    colstats_kernel<<<512, 256, 0, stream>>>(f32buf, mu2, var2, M, 512);
    bn_cast_kernel<<<(unsigned)(((size_t)M * 512 + 255) / 256), 256, 0, stream>>>(
        f32buf, mu2, var2, bn2_g, bn2_b, actA, (size_t)M * 512, 512);

    // ---- 18 head GEMMs: lrelu(l @ A_w[k] + A_b[k]) -> BN(axis=n) -> transpose
    for (int k = 0; k < NSc; ++k) {
        gemm(actA, Awp + (size_t)k * 512 * 512, A_b + (size_t)k * 512, f32buf, M, 512, 512, 1);
        colstats_kernel<<<512, 256, 0, stream>>>(f32buf, mu1, var1, M, 512);
        a_bn_transpose_kernel<<<(unsigned)(((size_t)M * 512 + 255) / 256), 256, 0, stream>>>(
            f32buf, mu1, var1, A_g + (size_t)k * 512, A_be + (size_t)k * 512,
            lw, k, NSc, 512, (size_t)M * 512);
    }
}
